// SupConSegLoss_850403524936
// MI455X (gfx1250) — matrix operand order repaired and checked by reading, not compiled, not run on hardware
//
#include <hip/hip_runtime.h>
#include <stdint.h>

// ---------------- problem constants ----------------
constexpr int Bc = 4, Vc = 2, Cc = 128, Hc = 32, Wc = 32;
constexpr int NTOT = 2 * Bc * Hc * Wc;            // 8192 anchors
constexpr int ROWS_PER_BLOCK = 128;               // 8 waves * 16 rows
constexpr int COLS_PER_TILE  = 64;                // LDS column tile
constexpr int SLICES         = 8;                 // column split for parallelism
constexpr int COLS_PER_SLICE = NTOT / SLICES;     // 1024
constexpr int NT_TILES       = COLS_PER_SLICE / COLS_PER_TILE; // 16
#define TEMP_INV (1.0f / 0.7f)

// ---------------- WMMA vector types ----------------
typedef __attribute__((ext_vector_type(16))) __bf16 v16bf;
typedef __attribute__((ext_vector_type(8)))  float  v8f;

union Frag {
    uint4 q[2];   // 32 bytes
    v16bf v;      // 16 bf16 elements
};

// round-to-nearest-even f32 -> bf16
static __device__ __forceinline__ unsigned short f2bf(float x) {
    unsigned u = __float_as_uint(x);
    unsigned r = (u + 0x7FFFu + ((u >> 16) & 1u)) >> 16;
    return (unsigned short)r;
}

// ---------------- kernel 1: pack features to bf16 [N][128], labels, zero ws --
__global__ void supcon_pack(const float* __restrict__ feats,
                            const int* __restrict__ labels,
                            unsigned short* __restrict__ packedF,
                            int* __restrict__ labG,
                            float* __restrict__ rowE,
                            float* __restrict__ rowP,
                            float* __restrict__ rowC,
                            float* __restrict__ accum) {
    int tid = blockIdx.x * blockDim.x + threadIdx.x;
    if (tid == 0) { accum[0] = 0.f; accum[1] = 0.f; }
    if (tid < NTOT) { rowE[tid] = 0.f; rowP[tid] = 0.f; rowC[tid] = 0.f; }
    if (tid >= NTOT * Cc) return;
    int n = tid >> 7;            // anchor index, (b', i, j) row-major
    int c = tid & 127;
    int bp  = n >> 10;           // b' in [0, 2B)
    int rem = n & 1023;
    int i = rem >> 5;
    int j = rem & 31;
    int v = (bp >= Bc) ? 1 : 0;  // concat([view0, view1]) on batch dim
    int b = bp - v * Bc;
    size_t fidx = ((((size_t)(b * Vc + v)) * Cc + c) * Hc + i) * Wc + j;
    packedF[(size_t)n * Cc + c] = f2bf(feats[fidx]);
    if (c == 0) {
        labG[n] = labels[(((size_t)(b * Vc + v)) * Hc + i) * Wc + j];
    }
}

// ---------------- kernel 2: fused Gram + per-row partial reductions ----------
// grid: 512 blocks = 64 row-blocks x 8 column slices
__global__ __launch_bounds__(256, 1)
void supcon_main(const unsigned short* __restrict__ packedF,
                 const int* __restrict__ labG,
                 float* __restrict__ rowE,
                 float* __restrict__ rowP,
                 float* __restrict__ rowC) {
    __shared__ uint4 ldsF[2][COLS_PER_TILE * 16];   // 2 x 16KB ping-pong
    __shared__ int   ldsLab[COLS_PER_SLICE];        // 4KB slice labels

    const int tid  = threadIdx.x;
    const int lane = tid & 31;
    const int wv   = tid >> 5;
    const int m    = lane & 15;       // N position / A row position
    const int hi   = lane >> 4;       // half-wave selector
    const int rb   = blockIdx.x >> 3;
    const int cs   = blockIdx.x & 7;
    const int rows_base  = rb * ROWS_PER_BLOCK + wv * 16;
    const int slice_base = cs * COLS_PER_SLICE;

    // --- A fragments for this wave's 16 rows (resident whole kernel) ---
    // 16-bit A 16x32 layout: lane m, elems 0..7 <- K = 32f + 8*hi + e,
    //                                elems 8..15 <- K = 32f + 16 + 8*hi + (e-8)
    Frag a[4];
    {
        const uint4* rq = (const uint4*)packedF + (size_t)(rows_base + m) * 16;
        #pragma unroll
        for (int f = 0; f < 4; ++f) {
            a[f].q[0] = rq[4 * f + hi];
            a[f].q[1] = rq[4 * f + 2 + hi];
        }
    }
    // labels of the 8 rows this lane's C-slots correspond to (M = v + 8*hi)
    int rowlab[8];
    #pragma unroll
    for (int v = 0; v < 8; ++v) rowlab[v] = labG[rows_base + 8 * hi + v];

    // stage this slice's labels once
    #pragma unroll
    for (int s = 0; s < COLS_PER_SLICE / 256; ++s)
        ldsLab[s * 256 + tid] = labG[slice_base + s * 256 + tid];

    // async (ASYNCcnt) DMA of one 16KB column tile directly into LDS:
    // 4 x global_load_async_to_lds_b128 per lane (64B/lane, 16KB/wave-group)
    auto issue_tile = [&](int buf, int tile) {
        const char* gsrc = (const char*)packedF
                         + (size_t)(slice_base + tile * COLS_PER_TILE) * 256;
        #pragma unroll
        for (int s = 0; s < 4; ++s) {
            uint32_t loff = (uint32_t)(size_t)(const void*)&ldsF[buf][s * 256 + tid];
            uint64_t ga   = (uint64_t)(size_t)(gsrc + s * 4096 + (size_t)tid * 16);
            asm volatile("global_load_async_to_lds_b128 %0, %1, off"
                         :: "v"(loff), "v"(ga) : "memory");
        }
    };

    issue_tile(0, 0);   // prologue: tile 0 in flight

    float esum[8], psum[8], csum[8];
    #pragma unroll
    for (int v = 0; v < 8; ++v) { esum[v] = 0.f; psum[v] = 0.f; csum[v] = 0.f; }

    for (int t = 0; t < NT_TILES; ++t) {
        // issue next tile's DMA, then wait until tile t is resident
        // (async loads complete in order: <=4 outstanding == tile t done)
        if (t + 1 < NT_TILES) {
            issue_tile((t + 1) & 1, t + 1);
            asm volatile("s_wait_asynccnt 0x4" ::: "memory");
        } else {
            asm volatile("s_wait_asynccnt 0x0" ::: "memory");
        }
        __syncthreads();

        const uint4* fbuf = ldsF[t & 1];
        const int ct = slice_base + t * COLS_PER_TILE;

        #pragma unroll
        for (int sub = 0; sub < 4; ++sub) {
            // B 32x16 bf16 layout: lane holds col n = m, K = 32f + 8*hi + e (e<8), 32f + 16 + 8*hi + (e-8) (e>=8)
            const int colq = (sub * 16 + m) * 16;
            Frag b[4];
            #pragma unroll
            for (int f = 0; f < 4; ++f) {
                int qi = colq + 4 * f + hi;
                b[f].q[0] = fbuf[qi];
                b[f].q[1] = fbuf[qi + 2];
            }
            v8f acc = {0.f, 0.f, 0.f, 0.f, 0.f, 0.f, 0.f, 0.f};
            #pragma unroll
            for (int f = 0; f < 4; ++f) {
                acc = __builtin_amdgcn_wmma_f32_16x16x32_bf16(
                    /*neg_a=*/false, a[f].v, /*neg_b=*/false, b[f].v,
                    /*c_mod=*/(short)0, acc, /*reuse_a=*/false, /*reuse_b=*/false);
            }
            const int colg   = ct + sub * 16 + m;       // this lane's column
            const int collab = ldsLab[t * COLS_PER_TILE + sub * 16 + m];
            #pragma unroll
            for (int v = 0; v < 8; ++v) {
                const int rowg = rows_base + 8 * hi + v; // C layout: M = v+8*hi
                float s  = acc[v] * TEMP_INV;
                float nd = (rowg != colg) ? 1.f : 0.f;   // exclude diagonal
                float sm = (rowlab[v] == collab) ? nd : 0.f;
                esum[v] += nd * __expf(s);
                psum[v] += sm * s;
                csum[v] += sm;
            }
        }
        __syncthreads();   // compute done before next iteration's DMA overwrite
    }

    // reduce the 16 column-lanes per half-wave (rows are per (hi, v))
    #pragma unroll
    for (int v = 0; v < 8; ++v) {
        #pragma unroll
        for (int off = 8; off >= 1; off >>= 1) {
            esum[v] += __shfl_xor(esum[v], off, 16);
            psum[v] += __shfl_xor(psum[v], off, 16);
            csum[v] += __shfl_xor(csum[v], off, 16);
        }
    }

    if (m == 0) {   // lanes 0 and 16: fold 8 rows each into global partials
        #pragma unroll
        for (int v = 0; v < 8; ++v) {
            const int row = rows_base + 8 * hi + v;
            atomicAdd(&rowE[row], esum[v]);
            atomicAdd(&rowP[row], psum[v]);
            atomicAdd(&rowC[row], csum[v]);
        }
    }
}

// ---------------- kernel 3: per-row loss + global sum ----------------
__global__ void supcon_rowloss(const int* __restrict__ labG,
                               const float* __restrict__ rowE,
                               const float* __restrict__ rowP,
                               const float* __restrict__ rowC,
                               float* __restrict__ accum) {
    int r = blockIdx.x * blockDim.x + threadIdx.x;
    float loss = 0.f, cnt = 0.f;
    if (r < NTOT) {
        int lab = labG[r];
        if (lab != 0) {
            loss = __logf(rowE[r]) - rowP[r] / rowC[r];
            cnt  = 1.f;
        }
    }
    #pragma unroll
    for (int off = 16; off >= 1; off >>= 1) {
        loss += __shfl_xor(loss, off, 32);
        cnt  += __shfl_xor(cnt,  off, 32);
    }
    if ((threadIdx.x & 31) == 0) {
        atomicAdd(&accum[0], loss);
        atomicAdd(&accum[1], cnt);
    }
}

// ---------------- kernel 4: finalize scalar ----------------
__global__ void supcon_finalize(const float* __restrict__ accum,
                                float* __restrict__ out) {
    float s = accum[0], c = accum[1];
    out[0] = (c > 0.f) ? s / fmaxf(c, 1.f) : 0.f;
}

// ---------------- launcher ----------------
extern "C" void kernel_launch(void* const* d_in, const int* in_sizes, int n_in,
                              void* d_out, int out_size, void* d_ws, size_t ws_size,
                              hipStream_t stream) {
    (void)in_sizes; (void)n_in; (void)out_size; (void)ws_size;
    const float* feats  = (const float*)d_in[0];   // [B,V,C,H,W] f32
    const int*   labels = (const int*)d_in[1];     // [B,V,H,W] i32

    char* ws = (char*)d_ws;
    size_t o = 0;
    unsigned short* packedF = (unsigned short*)(ws + o); o += (size_t)NTOT * Cc * 2; // 2MB
    int*   labG  = (int*)(ws + o);   o += (size_t)NTOT * 4;   // 32KB
    float* rowE  = (float*)(ws + o); o += (size_t)NTOT * 4;   // 32KB
    float* rowP  = (float*)(ws + o); o += (size_t)NTOT * 4;   // 32KB
    float* rowC  = (float*)(ws + o); o += (size_t)NTOT * 4;   // 32KB
    float* accum = (float*)(ws + o);

    supcon_pack<<<(NTOT * Cc + 255) / 256, 256, 0, stream>>>(
        feats, labels, packedF, labG, rowE, rowP, rowC, accum);
    supcon_main<<<(NTOT / ROWS_PER_BLOCK) * SLICES, 256, 0, stream>>>(
        packedF, labG, rowE, rowP, rowC);
    supcon_rowloss<<<NTOT / 256, 256, 0, stream>>>(labG, rowE, rowP, rowC, accum);
    supcon_finalize<<<1, 1, 0, stream>>>(accum, (float*)d_out);
}
